// MultiheadAttentionModule_42846593745002
// MI455X (gfx1250) — matrix operand order repaired and checked by reading, not compiled, not run on hardware
//
#include <hip/hip_runtime.h>
#include <hip/hip_bf16.h>
#include <cstdint>

// ---------------------------------------------------------------------------
// Problem constants (from reference): B=16, N=1024, DIM_IN=1024, D_FEAT=64, H=4
// ---------------------------------------------------------------------------
#define DB 16
#define DN 1024
#define DC 1024
#define DD 64
#define DH 4
// halves per one of Q/K/V region: 2 inputs * B * H * N * D
#define QKV_SZ ((size_t)2 * DB * DH * DN * DD)   // 8,388,608 halves

typedef _Float16 v16h __attribute__((ext_vector_type(16)));
typedef _Float16 h8   __attribute__((ext_vector_type(8)));
typedef float    v8f  __attribute__((ext_vector_type(8)));
typedef int      v4i  __attribute__((ext_vector_type(4)));

// ---------------------------------------------------------------------------
// Optional CDNA5 async global->LDS path (ASYNCcnt-tracked), guarded so the
// file compiles on toolchains without the builtins. Prototype (from hipcc
// diagnostic): (int4* global_src, int4* lds_dst, imm offset, imm cpol).
// ---------------------------------------------------------------------------
#if defined(__has_builtin)
#if __has_builtin(__builtin_amdgcn_global_load_async_to_lds_b128) && \
    __has_builtin(__builtin_amdgcn_s_wait_asynccnt)
#define HAS_ASYNC_LDS 1
#endif
#endif
#ifndef HAS_ASYNC_LDS
#define HAS_ASYNC_LDS 0
#endif

__device__ __forceinline__ void cp16_g2l(const _Float16* g, _Float16* l) {
#if HAS_ASYNC_LDS
  __builtin_amdgcn_global_load_async_to_lds_b128((v4i*)g, (v4i*)l, 0, 0);
#else
  *(h8*)l = *(const h8*)g;
#endif
}

__device__ __forceinline__ void async_fence() {
#if HAS_ASYNC_LDS
  __builtin_amdgcn_s_wait_asynccnt(0);
#endif
}

// ---------------------------------------------------------------------------
// WMMA helpers (CDNA5 v_wmma_f32_16x16x32_f16, wave32)
// ---------------------------------------------------------------------------
__device__ __forceinline__ v8f wmma16(v16h a, v16h b, v8f c) {
  // (neg_a, A, neg_b, B, c_mod, C, reuse_a, reuse_b)
  return __builtin_amdgcn_wmma_f32_16x16x32_f16(false, a, false, b, (short)0, c,
                                                false, false);
}

// A fragment (16x32 f16, M x K), storage row-major: base[row*ld + k].
// ISA layout: lanes 0-15 -> rows 0-15 holding K = {0..7, 16..23};
//             lanes 16-31 -> rows 0-15 holding K = {8..15, 24..31}.
__device__ __forceinline__ v16h load_a_frag(const _Float16* base, int ld, int lane) {
  const int row = lane & 15;
  const int kb  = (lane < 16) ? 0 : 8;
  const _Float16* p = base + row * ld + kb;
  h8 lo = *(const h8*)(p);
  h8 hi = *(const h8*)(p + 16);
  v16h a;
#pragma unroll
  for (int i = 0; i < 8; ++i) { a[i] = lo[i]; a[8 + i] = hi[i]; }
  return a;
}

// B fragment (32x16 f16, K x N), storage: column n's K-vector contiguous,
// i.e. element (k, n) at base[n*ld + k].
// ISA layout: lanes 0-15 -> cols 0-15 holding K = {0..7, 16..23};
//             lanes 16-31 -> cols 0-15 holding K = {8..15, 24..31}.
__device__ __forceinline__ v16h load_b_frag(const _Float16* base, int ld, int lane) {
  const int n  = lane & 15;
  const int kb = (lane < 16) ? 0 : 8;
  const _Float16* p = base + n * ld + kb;
  h8 lo = *(const h8*)(p);
  h8 hi = *(const h8*)(p + 16);
  v16h b;
#pragma unroll
  for (int i = 0; i < 8; ++i) { b[i] = lo[i]; b[8 + i] = hi[i]; }
  return b;
}

// ---------------------------------------------------------------------------
// Kernel 1: QKV projection GEMM.
//   out[input, matrix, b, h, n, d] = sum_c x[input][b,n,c] * W[matrix][h,d,c] + bias
// Tile: 128 rows x 64 cols per block, K-loop over 1024 in chunks of 64.
// f32 loads converted to f16 in LDS; fully unrolled staging loops.
// ---------------------------------------------------------------------------
__global__ __launch_bounds__(256) void qkv_proj_kernel(
    const float* __restrict__ xk, const float* __restrict__ xq,
    const float* __restrict__ Wq, const float* __restrict__ bq,
    const float* __restrict__ Wk, const float* __restrict__ bk,
    const float* __restrict__ Wv, const float* __restrict__ bv,
    _Float16* __restrict__ ws) {
  __shared__ _Float16 As[128][72];  // [row][k]   (BLK_K = 64, pad 8)
  __shared__ _Float16 Bs[64][72];   // [d][k]     (k contiguous -> B-frag friendly)

  const int tid  = threadIdx.x;
  const int lane = tid & 31;
  const int wid  = tid >> 5;
  const int wm   = wid >> 1;   // 0..3 -> 32-row strip
  const int wn   = wid & 1;    // 0..1 -> 32-col strip

  const int t      = blockIdx.x;   // 0..11
  const int matrix = t >> 2;       // 0=Q,1=K,2=V
  const int h      = t & 3;
  const int m0     = blockIdx.y * 128;   // global row (b*N + n)
  const int input  = blockIdx.z;

  const float* x    = input ? xq : xk;
  const float* W    = (matrix == 0) ? Wq : (matrix == 1) ? Wk : Wv;
  const float* bias = (matrix == 0) ? bq : (matrix == 1) ? bk : bv;

  v8f acc[2][2];
#pragma unroll
  for (int i = 0; i < 2; ++i)
#pragma unroll
    for (int j = 0; j < 2; ++j) acc[i][j] = (v8f)(0.0f);

  for (int kt = 0; kt < DC / 64; ++kt) {
    const int c0 = kt * 64;
    __syncthreads();
    // A tile: 128x64 f32 -> f16 (2048 float4 loads, 8 per thread, unrolled)
#pragma unroll
    for (int it = 0; it < 8; ++it) {
      const int idx = tid + it * 256;
      const int r   = idx >> 4;          // 16 float4 per 64-wide row
      const int cc  = (idx & 15) * 4;
      const float4 f = *(const float4*)(x + (size_t)(m0 + r) * DC + c0 + cc);
      As[r][cc + 0] = (_Float16)f.x;
      As[r][cc + 1] = (_Float16)f.y;
      As[r][cc + 2] = (_Float16)f.z;
      As[r][cc + 3] = (_Float16)f.w;
    }
    // B tile: 64x64 from W[h*64+d][c] (4 float4 per thread, unrolled)
#pragma unroll
    for (int it = 0; it < 4; ++it) {
      const int idx = tid + it * 256;
      const int d   = idx >> 4;
      const int cc  = (idx & 15) * 4;
      const float4 f = *(const float4*)(W + (size_t)(h * DD + d) * DC + c0 + cc);
      Bs[d][cc + 0] = (_Float16)f.x;
      Bs[d][cc + 1] = (_Float16)f.y;
      Bs[d][cc + 2] = (_Float16)f.z;
      Bs[d][cc + 3] = (_Float16)f.w;
    }
    // prefetch next A chunk (emits global_prefetch_b8)
    if (kt + 1 < DC / 64)
      __builtin_prefetch(x + (size_t)(m0 + (tid >> 1)) * DC + c0 + 64 + (tid & 1) * 32, 0, 1);
    __syncthreads();

    const v16h a00 = load_a_frag(&As[wm * 32 + 0][0], 72, lane);
    const v16h a01 = load_a_frag(&As[wm * 32 + 0][32], 72, lane);
    const v16h a10 = load_a_frag(&As[wm * 32 + 16][0], 72, lane);
    const v16h a11 = load_a_frag(&As[wm * 32 + 16][32], 72, lane);
    const v16h b00 = load_b_frag(&Bs[wn * 32 + 0][0], 72, lane);
    const v16h b01 = load_b_frag(&Bs[wn * 32 + 0][32], 72, lane);
    const v16h b10 = load_b_frag(&Bs[wn * 32 + 16][0], 72, lane);
    const v16h b11 = load_b_frag(&Bs[wn * 32 + 16][32], 72, lane);
    acc[0][0] = wmma16(a00, b00, acc[0][0]);
    acc[0][0] = wmma16(a01, b01, acc[0][0]);
    acc[0][1] = wmma16(a00, b10, acc[0][1]);
    acc[0][1] = wmma16(a01, b11, acc[0][1]);
    acc[1][0] = wmma16(a10, b00, acc[1][0]);
    acc[1][0] = wmma16(a11, b01, acc[1][0]);
    acc[1][1] = wmma16(a10, b10, acc[1][1]);
    acc[1][1] = wmma16(a11, b11, acc[1][1]);
  }

  // Epilogue: + bias, f16 store into ws region for this matrix
  const int b_ = m0 >> 10;        // batch
  const int n0 = m0 & (DN - 1);   // row within batch
  _Float16* outBase = ws + (size_t)matrix * QKV_SZ +
                      ((size_t)((input * DB + b_) * DH + h)) * DN * DD;
  const int hb      = (lane < 16) ? 0 : 8;
  const int colLane = lane & 15;
#pragma unroll
  for (int ni = 0; ni < 2; ++ni) {
    const int dcol = wn * 32 + ni * 16 + colLane;
    const float bb = bias[h * DD + dcol];
#pragma unroll
    for (int mi = 0; mi < 2; ++mi) {
#pragma unroll
      for (int j = 0; j < 8; ++j) {
        const int row = n0 + wm * 32 + mi * 16 + j + hb;
        outBase[(size_t)row * DD + dcol] = (_Float16)(acc[mi][ni][j] + bb);
      }
    }
  }
}

// ---------------------------------------------------------------------------
// Kernel 2: flash attention per (input, b, h). Each block handles 128 query
// rows (wave w owns rows w*16..w*16+15). Keys streamed in 64-row tiles.
// 1/sqrt(d) is folded into the Q fragments (exact power of two in f16).
// ---------------------------------------------------------------------------
__global__ __launch_bounds__(256) void attn_kernel(_Float16* __restrict__ ws) {
  __shared__ _Float16 Ks[64][72];        // [key][c]
  __shared__ _Float16 Vst[64][72];       // [d][key]  (transposed V tile)
  __shared__ _Float16 Ps[8][16][72];     // per-wave P staging (C-layout -> A-layout)

  const int tid  = threadIdx.x;
  const int lane = tid & 31;
  const int wid  = tid >> 5;
  const int qt    = blockIdx.x;   // 0..7
  const int bh    = blockIdx.y;   // 0..63
  const int input = blockIdx.z;   // 0=key_feat, 1=query_feat
  const int b_ = bh >> 2, h = bh & 3;

  const size_t headOff = ((size_t)((input * DB + b_) * DH + h)) * DN * DD;
  const _Float16* Qg = ws + headOff;
  const _Float16* Kg = ws + QKV_SZ + headOff;
  const _Float16* Vg = ws + 2 * QKV_SZ + headOff;
  _Float16*       Zg = ws + 3 * QKV_SZ + headOff;

  const int qrow0 = qt * 128 + wid * 16;

  // Q fragments (K = feature dim, chunks 0..31 / 32..63), pre-scaled by 1/8
  v16h qa0 = load_a_frag(Qg + (size_t)qrow0 * DD + 0, DD, lane);
  v16h qa1 = load_a_frag(Qg + (size_t)qrow0 * DD + 32, DD, lane);
#pragma unroll
  for (int i = 0; i < 16; ++i) {
    qa0[i] = qa0[i] * (_Float16)0.125f;
    qa1[i] = qa1[i] * (_Float16)0.125f;
  }

  float mrow[8], lrow[8];
  v8f accO[4];
#pragma unroll
  for (int j = 0; j < 8; ++j) { mrow[j] = -1e30f; lrow[j] = 0.0f; }
#pragma unroll
  for (int dt = 0; dt < 4; ++dt) accO[dt] = (v8f)(0.0f);

  const int hb      = (lane < 16) ? 0 : 8;
  const int colLane = lane & 15;

  for (int ktile = 0; ktile < DN / 64; ++ktile) {
    __syncthreads();
    // Load K tile (async global->LDS when available) and V tile (transposed)
#pragma unroll
    for (int it = 0; it < 2; ++it) {
      const int idx = tid + it * 256;
      const int key = idx >> 3;
      const int cc  = (idx & 7) * 8;
      cp16_g2l(Kg + (size_t)(ktile * 64 + key) * DD + cc, &Ks[key][cc]);
      const h8 v = *(const h8*)(Vg + (size_t)(ktile * 64 + key) * DD + cc);
#pragma unroll
      for (int j2 = 0; j2 < 8; ++j2) Vst[cc + j2][key] = v[j2];
    }
    async_fence();
    __syncthreads();

    // S = (Q/8) K^T for this wave's 16 rows x 64 keys (4 tiles of 16 keys)
    v8f s[4];
#pragma unroll
    for (int ni = 0; ni < 4; ++ni) {
      s[ni] = (v8f)(0.0f);
      const v16h kb0 = load_b_frag(&Ks[ni * 16][0], 72, lane);
      const v16h kb1 = load_b_frag(&Ks[ni * 16][32], 72, lane);
      s[ni] = wmma16(qa0, kb0, s[ni]);
      s[ni] = wmma16(qa1, kb1, s[ni]);
    }

    // Online softmax: per-row (j / j+8 split across lane halves) reductions
#pragma unroll
    for (int j = 0; j < 8; ++j) {
      float mx = fmaxf(fmaxf(s[0][j], s[1][j]), fmaxf(s[2][j], s[3][j]));
#pragma unroll
      for (int m = 1; m <= 8; m <<= 1) mx = fmaxf(mx, __shfl_xor(mx, m, 32));
      const float mnew  = fmaxf(mrow[j], mx);
      const float alpha = __expf(mrow[j] - mnew);
      float sum = 0.0f;
      float p[4];
#pragma unroll
      for (int ni = 0; ni < 4; ++ni) {
        p[ni] = __expf(s[ni][j] - mnew);
        sum += p[ni];
      }
#pragma unroll
      for (int m = 1; m <= 8; m <<= 1) sum += __shfl_xor(sum, m, 32);
      lrow[j] = lrow[j] * alpha + sum;
      mrow[j] = mnew;
#pragma unroll
      for (int dt = 0; dt < 4; ++dt) accO[dt][j] *= alpha;
      // stage P (C-layout) into LDS so it can be re-read in A-layout
#pragma unroll
      for (int ni = 0; ni < 4; ++ni)
        Ps[wid][j + hb][ni * 16 + colLane] = (_Float16)p[ni];
    }

    // O += P V   (A = P 16x64 over keys, B = V tile via transposed storage)
    const v16h pa0 = load_a_frag(&Ps[wid][0][0], 72, lane);
    const v16h pa1 = load_a_frag(&Ps[wid][0][32], 72, lane);
#pragma unroll
    for (int dt = 0; dt < 4; ++dt) {
      const v16h vb0 = load_b_frag(&Vst[dt * 16][0], 72, lane);
      const v16h vb1 = load_b_frag(&Vst[dt * 16][32], 72, lane);
      accO[dt] = wmma16(pa0, vb0, accO[dt]);
      accO[dt] = wmma16(pa1, vb1, accO[dt]);
    }
  }

  // Normalize and store Z (f16)
#pragma unroll
  for (int dt = 0; dt < 4; ++dt)
#pragma unroll
    for (int j = 0; j < 8; ++j) {
      const int row = qrow0 + j + hb;
      Zg[(size_t)row * DD + dt * 16 + colLane] = (_Float16)(accO[dt][j] / lrow[j]);
    }
}

// ---------------------------------------------------------------------------
// Kernel 3: corr = scale * Zq Zk^T per (b,h); output [b][q][h*N + k] in f32.
// K = 64 total -> no K loop: load both 128x64 tiles once, 16 WMMAs per wave.
// ---------------------------------------------------------------------------
__global__ __launch_bounds__(256) void corr_kernel(const _Float16* __restrict__ ws,
                                                   float* __restrict__ out) {
  __shared__ _Float16 Aq[128][72];
  __shared__ _Float16 Bk[128][72];

  const int tid  = threadIdx.x;
  const int lane = tid & 31;
  const int wid  = tid >> 5;
  const int wm = wid >> 1, wn = wid & 1;
  const int ktb = blockIdx.x, qtb = blockIdx.y;
  const int bh = blockIdx.z;
  const int b_ = bh >> 2, h = bh & 3;

  const _Float16* Zq = ws + 3 * QKV_SZ +
                       ((size_t)((1 * DB + b_) * DH + h)) * DN * DD;  // query_feat attn
  const _Float16* Zk = ws + 3 * QKV_SZ +
                       ((size_t)((0 * DB + b_) * DH + h)) * DN * DD;  // key_feat attn

#pragma unroll
  for (int it = 0; it < 4; ++it) {
    const int idx = tid + it * 256;
    const int r   = idx >> 3;
    const int cc  = (idx & 7) * 8;
    cp16_g2l(Zq + (size_t)(qtb * 128 + r) * DD + cc, &Aq[r][cc]);
    cp16_g2l(Zk + (size_t)(ktb * 128 + r) * DD + cc, &Bk[r][cc]);
  }
  async_fence();
  __syncthreads();

  v16h a[2][2];
  a[0][0] = load_a_frag(&Aq[wm * 32 + 0][0], 72, lane);
  a[0][1] = load_a_frag(&Aq[wm * 32 + 0][32], 72, lane);
  a[1][0] = load_a_frag(&Aq[wm * 32 + 16][0], 72, lane);
  a[1][1] = load_a_frag(&Aq[wm * 32 + 16][32], 72, lane);

  v8f acc[2][4];
#pragma unroll
  for (int mi = 0; mi < 2; ++mi)
#pragma unroll
    for (int ni = 0; ni < 4; ++ni) acc[mi][ni] = (v8f)(0.0f);

#pragma unroll
  for (int ni = 0; ni < 4; ++ni) {
    const v16h b0 = load_b_frag(&Bk[wn * 64 + ni * 16][0], 72, lane);
    const v16h b1 = load_b_frag(&Bk[wn * 64 + ni * 16][32], 72, lane);
#pragma unroll
    for (int mi = 0; mi < 2; ++mi) {
      acc[mi][ni] = wmma16(a[mi][0], b0, acc[mi][ni]);
      acc[mi][ni] = wmma16(a[mi][1], b1, acc[mi][ni]);
    }
  }

  const float scale   = 0.125f;
  const int   hb      = (lane < 16) ? 0 : 8;
  const int   colLane = lane & 15;
#pragma unroll
  for (int mi = 0; mi < 2; ++mi)
#pragma unroll
    for (int ni = 0; ni < 4; ++ni)
#pragma unroll
      for (int j = 0; j < 8; ++j) {
        const int qrow = qtb * 128 + wm * 32 + mi * 16 + j + hb;
        const int kcol = ktb * 128 + wn * 64 + ni * 16 + colLane;
        out[(size_t)(b_ * DN + qrow) * (DH * DN) + h * DN + kcol] =
            scale * acc[mi][ni][j];
      }
}

// ---------------------------------------------------------------------------
// Launch: d_in = {key_feat, query_feat, Wq, bq, Wk, bk, Wv, bv}
// Workspace: Q | K | V | Z regions of f16, 4 * 8,388,608 halves = 64 MB.
// ---------------------------------------------------------------------------
extern "C" void kernel_launch(void* const* d_in, const int* in_sizes, int n_in,
                              void* d_out, int out_size, void* d_ws, size_t ws_size,
                              hipStream_t stream) {
  (void)in_sizes; (void)n_in; (void)out_size; (void)ws_size;
  const float* key_feat   = (const float*)d_in[0];
  const float* query_feat = (const float*)d_in[1];
  const float* Wq = (const float*)d_in[2];
  const float* bq = (const float*)d_in[3];
  const float* Wk = (const float*)d_in[4];
  const float* bk = (const float*)d_in[5];
  const float* Wv = (const float*)d_in[6];
  const float* bv = (const float*)d_in[7];
  _Float16* ws  = (_Float16*)d_ws;
  float*    out = (float*)d_out;

  // 1) QKV projections: grid = (12 col-tiles, 128 row-tiles, 2 inputs)
  qkv_proj_kernel<<<dim3(12, 128, 2), 256, 0, stream>>>(
      key_feat, query_feat, Wq, bq, Wk, bk, Wv, bv, ws);
  // 2) Flash attention: grid = (8 q-tiles, B*H=64, 2 inputs)
  attn_kernel<<<dim3(8, 64, 2), 256, 0, stream>>>(ws);
  // 3) Cross-correlation + output transpose: grid = (8 k-tiles, 8 q-tiles, B*H=64)
  corr_kernel<<<dim3(8, 8, 64), 256, 0, stream>>>(ws, out);
}
